// MambaBlock_41154376630317
// MI455X (gfx1250) — matrix operand order repaired and checked by reading, not compiled, not run on hardware
//
#include <hip/hip_runtime.h>

// ---------------------------------------------------------------------------
// Mamba block for MI455X (gfx1250, wave32, WMMA + TDM).
//   1. cvt weights fp32->bf16
//   2. layernorm -> xn bf16
//   3. GEMM1  xz = xn @ W_in^T        (32768x256x1024)  wmma bf16, TDM A-tiles
//   4. depthwise causal conv(4)+SiLU -> xc bf16
//   5. GEMM2  x_dbl = xc @ W_xproj^T  (32768x512x144)
//   6. delta = softplus(dt @ W_dt^T + b_dt)
//   7. selective scan (1 wave per (b,d), shfl_xor reduce, fused gating)
//   8. GEMM3  pre = y @ W_out^T       (32768x512x256)
//   9. transpose + residual -> out
// ---------------------------------------------------------------------------

#define BSZ     8
#define DMODEL  256
#define DINNER  512
#define DSTATE  64
#define LSEQ    4096            // FDIM*TDIM
#define MROWS   (BSZ * LSEQ)    // 32768

typedef __bf16 v16bf __attribute__((ext_vector_type(16)));
typedef __bf16 v8bf  __attribute__((ext_vector_type(8)));
typedef float  v8f   __attribute__((ext_vector_type(8)));
typedef unsigned v4u __attribute__((ext_vector_type(4)));
typedef unsigned v8u __attribute__((ext_vector_type(8)));

static __device__ __forceinline__ __bf16 f2bf(float f) {
  unsigned u = __builtin_bit_cast(unsigned, f);
  u += 0x7FFFu + ((u >> 16) & 1u);
  unsigned short s = (unsigned short)(u >> 16);
  return __builtin_bit_cast(__bf16, s);
}
static __device__ __forceinline__ float bf2f(__bf16 b) {
  unsigned short s = __builtin_bit_cast(unsigned short, b);
  unsigned u = ((unsigned)s) << 16;
  return __builtin_bit_cast(float, u);
}
static __device__ __forceinline__ float silu(float v) {
  return v / (1.f + __expf(-v));
}

// ---------------- TDM: async load of a 16x32 bf16 tile into LDS ------------
// D# group0: [1:0]=count=1, [63:32]=lds_addr, [120:64]=global_addr, [127:126]=2
// D# group1: [17:16]=data_size(1=2B), [79:48]=tensor_dim0=K,
//            [111:80]=tensor_dim1, [127:112]=tile_dim0=32, [143:128]=tile_dim1=16,
//            [207:160]=tensor_dim0_stride=K (elements)
static __device__ __forceinline__ void tdm_load_a16x32(const __bf16* gptr,
                                                       unsigned lds_off,
                                                       unsigned K) {
  unsigned long long ga = (unsigned long long)(size_t)gptr;
  v4u g0;
  g0[0] = 1u;                                   // count=1
  g0[1] = lds_off;                              // lds_addr
  g0[2] = (unsigned)ga;                         // global_addr lo
  g0[3] = (unsigned)(ga >> 32) | (2u << 30);    // global_addr hi | type=2
  v8u g1;
  g1[0] = 1u << 16;                             // data_size = 2 bytes
  g1[1] = (K & 0xFFFFu) << 16;                  // tensor_dim0 lo16
  g1[2] = (K >> 16);                            // tensor_dim0 hi | dim1 lo16(=0)
  g1[3] = 0x10u | (32u << 16);                  // tensor_dim1 hi (1<<20) | tile_dim0=32
  g1[4] = 16u;                                  // tile_dim1 = 16 rows
  g1[5] = K;                                    // tensor_dim0_stride lo32
  g1[6] = 0u;
  g1[7] = 0u;
  asm volatile("tensor_load_to_lds %0, %1" :: "s"(g0), "s"(g1) : "memory");
}

// ------------------------- weight conversion -------------------------------
__global__ void cvt_f32_bf16(const float* __restrict__ in,
                             __bf16* __restrict__ out, int n) {
  int i = blockIdx.x * 256 + threadIdx.x;
  if (i < n) out[i] = f2bf(in[i]);
}

// ------------------------------ layernorm ----------------------------------
__global__ void ln_kernel(const float* __restrict__ x,
                          const float* __restrict__ gamma,
                          const float* __restrict__ beta,
                          __bf16* __restrict__ xn) {
  __shared__ float tile[32 * 257];
  __shared__ float psum[8][32];
  __shared__ float psq[8][32];
  __shared__ float smu[32], srs[32];

  int bid = blockIdx.x;              // B * 128
  int b   = bid >> 7;
  int l0  = (bid & 127) * 32;
  int tid = threadIdx.x;             // 256
  int lx  = tid & 31, cg = tid >> 5;

  const float* xb = x + (size_t)b * DMODEL * LSEQ;
  float s = 0.f, sq = 0.f;
  #pragma unroll 4
  for (int cc = 0; cc < 32; ++cc) {
    int c = cc * 8 + cg;
    float v = xb[(size_t)c * LSEQ + l0 + lx];
    tile[lx * 257 + c] = v;
    s += v; sq += v * v;
  }
  psum[cg][lx] = s; psq[cg][lx] = sq;
  __syncthreads();
  if (tid < 32) {
    float S = 0.f, Q = 0.f;
    #pragma unroll
    for (int g = 0; g < 8; ++g) { S += psum[g][tid]; Q += psq[g][tid]; }
    float mu  = S * (1.f / DMODEL);
    float var = Q * (1.f / DMODEL) - mu * mu;
    smu[tid] = mu;
    srs[tid] = rsqrtf(var + 1e-5f);
  }
  __syncthreads();
  float gm = gamma[tid], bt = beta[tid];
  #pragma unroll 4
  for (int l = 0; l < 32; ++l) {
    float v = tile[l * 257 + tid];
    float r = (v - smu[l]) * srs[l] * gm + bt;
    xn[((size_t)(b * LSEQ + l0 + l)) * DMODEL + tid] = f2bf(r);
  }
}

// ------------------------------ WMMA GEMM ----------------------------------
// One K-step: batch NT B-frag loads, wait for the A tile's TDM transfer,
// read the A frag from LDS, then issue NT back-to-back wmmas.
template <int NT, bool LAST>
static __device__ __forceinline__ void gemm_step(const __bf16* __restrict__ W,
                                                 int K, int nc0, int l16,
                                                 int half, int k0,
                                                 const __bf16* abuf,
                                                 v8f (&acc)[NT]) {
  v16bf bfr[NT];
  #pragma unroll
  for (int j = 0; j < NT; ++j) {
    const __bf16* wrow = W + (size_t)(nc0 + j * 16 + l16) * K + k0 + half * 16;
    bfr[j] = *(const v16bf*)wrow;
  }
  if (LAST) __builtin_amdgcn_s_wait_tensorcnt(0);
  else      __builtin_amdgcn_s_wait_tensorcnt(1);
  asm volatile("" ::: "memory");
  // A frag: lane<16 -> K {0..15}; lane>=16 -> {16..31}
  const __bf16* ap = abuf + l16 * 32 + half * 16;
  v8bf alo = *(const v8bf*)ap;
  v8bf ahi = *(const v8bf*)(ap + 8);
  v16bf af;
  #pragma unroll
  for (int i = 0; i < 8; ++i) { af[i] = alo[i]; af[i + 8] = ahi[i]; }
  #pragma unroll
  for (int j = 0; j < NT; ++j)
    acc[j] = __builtin_amdgcn_wmma_f32_16x16x32_bf16(
        false, af, false, bfr[j], (short)0, acc[j], false, false);
}

// C[M,N] = A[M,K](bf16 rowmajor) @ W[N,K](bf16 rowmajor)^T, f32 accumulate.
// WV waves/block; wave computes 16 x (16*NT). The A tile (16x32) streams
// through the Tensor Data Mover into per-wave double-buffered LDS (no
// barriers: each wave tracks its own TENSORcnt). Steady-state loop always
// prefetches the next slab; last K-step peeled so the loop is branch-free.
template <int WV, int NT, bool OUT_BF16>
__global__ void gemm_wmma(const __bf16* __restrict__ A,
                          const __bf16* __restrict__ W,
                          void* __restrict__ Cp, int M, int N, int K) {
  __shared__ alignas(16) __bf16 atile[WV][2][16 * 32];

  int lane = threadIdx.x & 31;
  int wv   = threadIdx.x >> 5;
  int Mt   = M / 16;
  int mi   = blockIdx.x % Mt;
  int nb   = blockIdx.x / Mt;
  int nc0  = nb * (WV * NT * 16) + wv * NT * 16;
  int half = lane >> 4;
  int l16  = lane & 15;

  const __bf16* abase = A + (size_t)mi * 16 * K;
  unsigned lds0 = (unsigned)(size_t)&atile[wv][0][0];
  unsigned lds1 = (unsigned)(size_t)&atile[wv][1][0];

  v8f acc[NT];
  #pragma unroll
  for (int j = 0; j < NT; ++j)
    #pragma unroll
    for (int r = 0; r < 8; ++r) acc[j][r] = 0.f;

  tdm_load_a16x32(abase, lds0, (unsigned)K);

  int cur = 0;
  int k0 = 0;
  for (; k0 + 32 < K; k0 += 32) {          // steady state: always prefetch
    tdm_load_a16x32(abase + k0 + 32, cur ? lds0 : lds1, (unsigned)K);
    __builtin_prefetch((const void*)(W + (size_t)(nc0 + l16) * K + k0 + 32),
                       0, 1);
    gemm_step<NT, false>(W, K, nc0, l16, half, k0, &atile[wv][cur][0], acc);
    cur ^= 1;
  }
  gemm_step<NT, true>(W, K, nc0, l16, half, k0, &atile[wv][cur][0], acc);

  int mrow0 = mi * 16 + half * 8;
  #pragma unroll
  for (int j = 0; j < NT; ++j) {
    int ncol = nc0 + j * 16 + l16;
    #pragma unroll
    for (int r = 0; r < 8; ++r) {
      size_t idx = (size_t)(mrow0 + r) * N + ncol;
      if constexpr (OUT_BF16) ((__bf16*)Cp)[idx] = f2bf(acc[j][r]);
      else                    ((float*)Cp)[idx]  = acc[j][r];
    }
  }
}

// --------------------- depthwise causal conv + SiLU ------------------------
__global__ void conv_silu_kernel(const __bf16* __restrict__ xz,
                                 const float* __restrict__ Wc,
                                 const float* __restrict__ bc,
                                 __bf16* __restrict__ xc) {
  int tid = threadIdx.x;
  int row = blockIdx.x;
  int b = row >> 12, l = row & 4095;
  #pragma unroll
  for (int p = 0; p < 2; ++p) {
    int d = tid + p * 256;
    float acc = bc[d];
    #pragma unroll
    for (int j = 0; j < 4; ++j) {
      int ls = l + j - 3;
      if (ls >= 0)
        acc += Wc[d * 4 + j] * bf2f(xz[((size_t)(b * 4096 + ls)) * 1024 + d]);
    }
    xc[(size_t)row * 512 + d] = f2bf(silu(acc));
  }
}

// ----------------------- delta = softplus(dt@W_dt^T+b) ---------------------
__global__ void delta_kernel(const float* __restrict__ xdbl,
                             const float* __restrict__ Wdt,
                             const float* __restrict__ bdt,
                             float* __restrict__ delta) {
  int tid = threadIdx.x;
  int row = blockIdx.x;
  const float* dt = xdbl + (size_t)row * 144;
  #pragma unroll
  for (int p = 0; p < 2; ++p) {
    int d = tid + p * 256;
    float acc = bdt[d];
    #pragma unroll
    for (int r = 0; r < 16; ++r) acc += dt[r] * Wdt[d * 16 + r];
    float sp = (acc > 20.f) ? acc : __logf(1.f + __expf(acc));
    delta[(size_t)row * 512 + d] = sp;
  }
}

// ----------------------------- selective scan ------------------------------
__global__ void scan_kernel(const float* __restrict__ xdbl,
                            const float* __restrict__ delta,
                            const __bf16* __restrict__ xc,
                            const __bf16* __restrict__ xz,
                            const float* __restrict__ A_log,
                            const float* __restrict__ Dsk,
                            __bf16* __restrict__ y) {
  int lane = threadIdx.x & 31;
  int wv   = threadIdx.x >> 5;
  int blk  = blockIdx.x;
  int b    = blk >> 6;
  int d    = (blk & 63) * 8 + wv;

  float A0 = -__expf(A_log[d * DSTATE + lane]);
  float A1 = -__expf(A_log[d * DSTATE + lane + 32]);
  float Dd = Dsk[d];
  float h0 = 0.f, h1 = 0.f;
  size_t rowbase = (size_t)b * LSEQ;

  for (int t = 0; t < LSEQ; ++t) {
    size_t row = rowbase + t;
    float dl = delta[row * 512 + d];
    float u  = bf2f(xc[row * 512 + d]);
    const float* xd = xdbl + row * 144;
    float B0 = xd[16 + lane], B1 = xd[48 + lane];
    float C0 = xd[80 + lane], C1 = xd[112 + lane];
    float du = dl * u;
    h0 = __expf(dl * A0) * h0 + du * B0;
    h1 = __expf(dl * A1) * h1 + du * B1;
    float yp = h0 * C0 + h1 * C1;
    #pragma unroll
    for (int m = 16; m >= 1; m >>= 1) yp += __shfl_xor(yp, m, 32);
    if (lane == 0) {
      float zv  = bf2f(xz[row * 1024 + 512 + d]);
      float out = (yp + u * Dd) * silu(zv);
      y[row * 512 + d] = f2bf(out);
    }
  }
}

// ----------------------- residual + transpose writeout ---------------------
__global__ void writeout_kernel(const float* __restrict__ pre,
                                const float* __restrict__ x,
                                float* __restrict__ out) {
  __shared__ float t[32][33];
  int bid = blockIdx.x;
  int b   = bid >> 10;
  int rem = bid & 1023;
  int l0  = (rem >> 3) * 32;
  int c0  = (rem & 7) * 32;
  int tid = threadIdx.x;
  int jj = tid & 31, ii = tid >> 5;
  #pragma unroll
  for (int s = 0; s < 4; ++s) {
    int i = ii + s * 8;
    t[i][jj] = pre[((size_t)(b * LSEQ + l0 + i)) * DMODEL + c0 + jj];
  }
  __syncthreads();
  #pragma unroll
  for (int s = 0; s < 4; ++s) {
    int i = ii + s * 8;
    size_t idx = ((size_t)(b * DMODEL + c0 + i)) * LSEQ + l0 + jj;
    out[idx] = t[jj][i] + x[idx];
  }
}

// ------------------------------- launcher ----------------------------------
extern "C" void kernel_launch(void* const* d_in, const int* in_sizes, int n_in,
                              void* d_out, int out_size, void* d_ws,
                              size_t ws_size, hipStream_t stream) {
  (void)in_sizes; (void)n_in; (void)out_size; (void)ws_size;
  const float* x       = (const float*)d_in[0];
  const float* gamma   = (const float*)d_in[1];
  const float* beta    = (const float*)d_in[2];
  const float* W_in    = (const float*)d_in[3];
  const float* W_conv  = (const float*)d_in[4];
  const float* b_conv  = (const float*)d_in[5];
  const float* W_xproj = (const float*)d_in[6];
  const float* W_dt    = (const float*)d_in[7];
  const float* b_dt    = (const float*)d_in[8];
  const float* A_log   = (const float*)d_in[9];
  const float* D_skip  = (const float*)d_in[10];
  const float* W_out   = (const float*)d_in[11];
  float* out = (float*)d_out;

  char* ws = (char*)d_ws;
  size_t off = 0;
  auto alloc = [&](size_t bytes) -> char* {
    char* p = ws + off;
    off += (bytes + 255) & ~(size_t)255;
    return p;
  };
  __bf16* xn    = (__bf16*)alloc((size_t)MROWS * 256 * 2);
  __bf16* xz    = (__bf16*)alloc((size_t)MROWS * 1024 * 2);
  __bf16* xc    = (__bf16*)alloc((size_t)MROWS * 512 * 2);
  float*  xdbl  = (float*) alloc((size_t)MROWS * 144 * 4);
  float*  dl    = (float*) alloc((size_t)MROWS * 512 * 4);
  __bf16* yb    = (__bf16*)alloc((size_t)MROWS * 512 * 2);
  float*  pre   = (float*) alloc((size_t)MROWS * 256 * 4);
  __bf16* winb  = (__bf16*)alloc((size_t)1024 * 256 * 2);
  __bf16* wxpb  = (__bf16*)alloc((size_t)144 * 512 * 2);
  __bf16* woutb = (__bf16*)alloc((size_t)256 * 512 * 2);

  cvt_f32_bf16<<<(1024 * 256 + 255) / 256, 256, 0, stream>>>(W_in, winb, 1024 * 256);
  cvt_f32_bf16<<<(144 * 512 + 255) / 256, 256, 0, stream>>>(W_xproj, wxpb, 144 * 512);
  cvt_f32_bf16<<<(256 * 512 + 255) / 256, 256, 0, stream>>>(W_out, woutb, 256 * 512);

  ln_kernel<<<BSZ * (LSEQ / 32), 256, 0, stream>>>(x, gamma, beta, xn);

  { // GEMM1: 32768x256 -> 1024; 4 waves x (16x64), 4 col-blocks
    int Mt = MROWS / 16, cb = 1024 / (4 * 4 * 16);
    gemm_wmma<4, 4, true><<<Mt * cb, 128, 0, stream>>>(xn, winb, xz,
                                                       MROWS, 1024, 256);
  }
  conv_silu_kernel<<<MROWS, 256, 0, stream>>>(xz, W_conv, b_conv, xc);
  { // GEMM2: 32768x512 -> 144; 3 waves x (16x48), 1 col-block
    int Mt = MROWS / 16;
    gemm_wmma<3, 3, false><<<Mt, 96, 0, stream>>>(xc, wxpb, xdbl,
                                                  MROWS, 144, 512);
  }
  delta_kernel<<<MROWS, 256, 0, stream>>>(xdbl, W_dt, b_dt, dl);
  scan_kernel<<<BSZ * (DINNER / 8), 256, 0, stream>>>(xdbl, dl, xc, xz, A_log,
                                                      D_skip, yb);
  { // GEMM3: 32768x512 -> 256; 4 waves x (16x64), 1 col-block
    int Mt = MROWS / 16;
    gemm_wmma<4, 4, false><<<Mt, 128, 0, stream>>>(yb, woutb, pre,
                                                   MROWS, 256, 512);
  }
  writeout_kernel<<<BSZ * 128 * 8, 256, 0, stream>>>(pre, x, out);
}
